// ISD_18614388261513
// MI455X (gfx1250) — matrix operand order repaired and checked by reading, not compiled, not run on hardware
//
#include <hip/hip_runtime.h>
#include <hip/hip_bf16.h>
#include <math.h>

// ---------------- problem constants (from reference) ----------------
#define BQ   256        // batch
#define DIM  512        // feature dim
#define KQ   128000     // queue size
#define TOPK 5
// scales: 1/T_S = 10, 1/T_T = 100

typedef __attribute__((ext_vector_type(16))) _Float16 v16h;
typedef __attribute__((ext_vector_type(8)))  _Float16 v8h;
typedef __attribute__((ext_vector_type(8)))  float    v8f;
typedef __attribute__((ext_vector_type(4)))  float    v4f;   // true vector type for NT loads

union V16 {
    v16h v;
    v8h  h8[2];
};

// ------------------------------------------------------------------
// Kernel 1: L2-normalize q (rows 0..255) and k (rows 256..511) into a
// stacked f16 A matrix [512 x 512] in workspace.
// ------------------------------------------------------------------
__global__ void __launch_bounds__(256)
ISD_moco_norm(const float* __restrict__ q, const float* __restrict__ k,
              _Float16* __restrict__ Ah)
{
    __shared__ float red[256];
    const int r   = blockIdx.x;           // 0..511
    const int tid = threadIdx.x;          // 0..255 (covers DIM=512 as 2 elems)
    const float* src = (r < BQ) ? (q + (size_t)r * DIM)
                                : (k + (size_t)(r - BQ) * DIM);
    float x0 = src[tid];
    float x1 = src[tid + 256];
    red[tid] = x0 * x0 + x1 * x1;
    __syncthreads();
    for (int off = 128; off >= 1; off >>= 1) {
        if (tid < off) red[tid] += red[tid + off];
        __syncthreads();
    }
    const float n   = sqrtf(red[0]);
    const float inv = 1.0f / fmaxf(n, 1e-12f);
    Ah[(size_t)r * DIM + tid]       = (_Float16)(x0 * inv);
    Ah[(size_t)r * DIM + tid + 256] = (_Float16)(x1 * inv);
}

// ------------------------------------------------------------------
// Kernel 2: C[512 x 128000] = A[512 x 512] * queue^T, f16 WMMA, f32 acc.
// One workgroup per 16-wide N tile (8000 blocks, 8 waves each).
// Rows 0..255 scaled by 10  -> sim_q region of d_out
// Rows 256..511 scaled by 100 -> sim_k region (same flat index formula).
// ------------------------------------------------------------------
#define LSTR 528   // LDS row stride in halves (1056 B: 32B aligned, bank-spread)

__global__ void __launch_bounds__(256)
ISD_moco_gemm(const _Float16* __restrict__ A,
              const float* __restrict__ queue,
              float* __restrict__ out)
{
    __shared__ _Float16 qtile[16 * LSTR];

    const int nbase = blockIdx.x * 16;    // queue-row tile base (N)
    const int tid   = threadIdx.x;

    // Stage 16 x 512 queue tile into LDS, converting f32 -> f16.
    // Thread t: tile row t>>4, 32 consecutive columns starting (t&15)*32.
    {
        const int r  = tid >> 4;
        const int c0 = (tid & 15) * 32;
        const float*   src = queue + (size_t)(nbase + r) * DIM + c0;
        _Float16*      dst = qtile + r * LSTR + c0;
#pragma unroll
        for (int i = 0; i < 8; ++i) {
            v4f f = __builtin_nontemporal_load((const v4f*)(src + i * 4));
            dst[i * 4 + 0] = (_Float16)f.x;
            dst[i * 4 + 1] = (_Float16)f.y;
            dst[i * 4 + 2] = (_Float16)f.z;
            dst[i * 4 + 3] = (_Float16)f.w;
        }
    }
    __syncthreads();

    const int wave = tid >> 5;            // 0..7  -> M rows [wave*64, wave*64+63]
    const int lane = tid & 31;
    const int lrow = lane & 15;
    const int sel  = lane >> 4;           // which K-half group this lane holds

    v8f acc[4] = {};                      // 4 M-tiles of 16x16 f32

    const _Float16* Abase = A + (size_t)(wave * 64 + lrow) * DIM;

    for (int kk = 0; kk < DIM; kk += 32) {
        // B fragment (32x16 f16): lane holds queue-tile row `lrow`,
        // 16 contiguous K values selected by `sel`.
        V16 b;
        const _Float16* bp = qtile + lrow * LSTR + kk + sel * 16;
        b.h8[0] = *(const v8h*)(bp);
        b.h8[1] = *(const v8h*)(bp + 8);

#pragma unroll
        for (int t = 0; t < 4; ++t) {
            // A fragment (16x32 f16): lane holds row M = tile_base + lrow,
            // K halves {0-15} (sel=0) or {16-31} (sel=1).
            V16 a;
            const _Float16* ap = Abase + (size_t)t * 16 * DIM + kk + sel * 16;
            a.h8[0] = *(const v8h*)(ap);
            a.h8[1] = *(const v8h*)(ap + 8);

            acc[t] = __builtin_amdgcn_wmma_f32_16x16x32_f16(
                /*neg_a=*/false, a.v, /*neg_b=*/false, b.v,
                /*c_mod=*/(short)0, acc[t],
                /*reuse_a=*/false, /*reuse_b=*/false);
        }
    }

    // Store: 16x16 f32 C layout -> VGPR r holds M = r (lanes 0-15) / r+8 (16-31).
#pragma unroll
    for (int t = 0; t < 4; ++t) {
        const int mbase = wave * 64 + t * 16 + sel * 8;
        const float scale = (mbase < BQ) ? 10.0f : 100.0f;  // 1/T_S vs 1/T_T
#pragma unroll
        for (int r = 0; r < 8; ++r) {
            const int m = mbase + r;
            __builtin_nontemporal_store(acc[t][r] * scale,
                                        out + (size_t)m * KQ + nbase + lrow);
        }
    }
}

// ------------------------------------------------------------------
// Kernel 3: per-row top-5 of sim_k, gather queue_labels, count matches.
// One block per batch row.
// ------------------------------------------------------------------
__global__ void __launch_bounds__(256)
ISD_moco_topk(const float* __restrict__ simk,
              const int* __restrict__ labels,
              const int* __restrict__ queue_labels,
              float* __restrict__ match_out)
{
    __shared__ float sv[256 * TOPK];
    __shared__ int   si[256 * TOPK];
    const int row = blockIdx.x;
    const int tid = threadIdx.x;
    const float* p = simk + (size_t)row * KQ;

    float tv[TOPK];
    int   ti[TOPK];
#pragma unroll
    for (int x = 0; x < TOPK; ++x) { tv[x] = -__builtin_inff(); ti[x] = 0; }

    for (int j = tid; j < KQ; j += 256) {
        const float v = p[j];
        if (v > tv[TOPK - 1]) {
            int pos = TOPK - 1;
            while (pos > 0 && v > tv[pos - 1]) {
                tv[pos] = tv[pos - 1]; ti[pos] = ti[pos - 1]; --pos;
            }
            tv[pos] = v; ti[pos] = j;
        }
    }
#pragma unroll
    for (int x = 0; x < TOPK; ++x) { sv[tid * TOPK + x] = tv[x]; si[tid * TOPK + x] = ti[x]; }

    for (int off = 128; off >= 1; off >>= 1) {
        __syncthreads();
        if (tid < off) {
            float va[TOPK], vb[TOPK]; int ia[TOPK], ib[TOPK];
#pragma unroll
            for (int x = 0; x < TOPK; ++x) {
                va[x] = sv[tid * TOPK + x];         ia[x] = si[tid * TOPK + x];
                vb[x] = sv[(tid + off) * TOPK + x]; ib[x] = si[(tid + off) * TOPK + x];
            }
            int pa = 0, pb = 0;
#pragma unroll
            for (int j = 0; j < TOPK; ++j) {
                const bool takeA = (pb >= TOPK) || (pa < TOPK && va[pa] >= vb[pb]);
                if (takeA) { sv[tid * TOPK + j] = va[pa]; si[tid * TOPK + j] = ia[pa]; ++pa; }
                else       { sv[tid * TOPK + j] = vb[pb]; si[tid * TOPK + j] = ib[pb]; ++pb; }
            }
        }
    }
    __syncthreads();

    if (tid == 0) {
        const int lab = labels[row];
        float cnt = 0.0f;
#pragma unroll
        for (int x = 0; x < TOPK; ++x)
            cnt += (queue_labels[si[x]] == lab) ? 1.0f : 0.0f;
        match_out[row] = cnt;
    }
}

// ------------------------------------------------------------------
// Kernel 4: purity = mean(matches / TOPK) over 256 rows.
// ------------------------------------------------------------------
__global__ void __launch_bounds__(256)
ISD_moco_purity(const float* __restrict__ matches, float* __restrict__ out_scalar)
{
    __shared__ float red[256];
    const int tid = threadIdx.x;
    red[tid] = matches[tid];
    __syncthreads();
    for (int off = 128; off >= 1; off >>= 1) {
        if (tid < off) red[tid] += red[tid + off];
        __syncthreads();
    }
    if (tid == 0) *out_scalar = red[0] / (float)(TOPK * BQ);
}

// ------------------------------------------------------------------
extern "C" void kernel_launch(void* const* d_in, const int* in_sizes, int n_in,
                              void* d_out, int out_size, void* d_ws, size_t ws_size,
                              hipStream_t stream)
{
    const float* q            = (const float*)d_in[0];
    const float* k            = (const float*)d_in[1];
    const float* queue        = (const float*)d_in[2];
    const int*   labels       = (const int*)d_in[3];
    const int*   queue_labels = (const int*)d_in[4];
    float*       out          = (float*)d_out;

    // workspace layout: A_h [512*512 f16] (512 KB), then match counts [256 f32]
    _Float16* Ah      = (_Float16*)d_ws;
    float*    matches = (float*)((char*)d_ws + (size_t)(2 * BQ) * DIM * sizeof(_Float16));

    // 1) normalize q,k -> stacked f16 A
    ISD_moco_norm<<<2 * BQ, 256, 0, stream>>>(q, k, Ah);

    // 2) fused dual GEMM (sim_q rows 0..255, sim_k rows 256..511 of d_out)
    ISD_moco_gemm<<<KQ / 16, 256, 0, stream>>>(Ah, queue, out);

    // 3) top-5 over sim_k (second half of d_out) -> per-row match counts
    ISD_moco_topk<<<BQ, 256, 0, stream>>>(out + (size_t)BQ * KQ, labels,
                                          queue_labels, matches);

    // 4) purity scalar at d_out[2*B*K]
    ISD_moco_purity<<<1, 256, 0, stream>>>(matches, out + (size_t)2 * BQ * KQ);
}
